// StructuralGNN_53541062312062
// MI455X (gfx1250) — matrix operand order repaired and checked by reading, not compiled, not run on hardware
//
#include <hip/hip_runtime.h>
#include <hip/hip_bf16.h>

typedef __attribute__((ext_vector_type(16))) _Float16 v16h;
typedef __attribute__((ext_vector_type(8)))  _Float16 v8h;
typedef __attribute__((ext_vector_type(8)))  float    v8f;

#define D_H   128
#define HEADS 2

// ---------------------------------------------------------------------------
// utility: grid-stride fill
// ---------------------------------------------------------------------------
__global__ void fill_f32(float* __restrict__ p, float v, size_t n) {
  size_t i = (size_t)blockIdx.x * blockDim.x + threadIdx.x;
  size_t stride = (size_t)gridDim.x * blockDim.x;
  for (; i < n; i += stride) p[i] = v;
}

// ---------------------------------------------------------------------------
// degree / norm
// ---------------------------------------------------------------------------
__global__ void deg_kernel(const int* __restrict__ dst, float* __restrict__ deg,
                           long long E) {
  long long t = (long long)blockIdx.x * blockDim.x + threadIdx.x;
  if (t < E) atomicAdd(&deg[dst[t]], 1.0f);
}

__global__ void dinv_kernel(const float* __restrict__ deg, float* __restrict__ dinv,
                            long long N) {
  long long t = (long long)blockIdx.x * blockDim.x + threadIdx.x;
  if (t < N) dinv[t] = rsqrtf(deg[t] + 1.0f);   // +1 self loop; always > 0
}

// ---------------------------------------------------------------------------
// WMMA GEMM:  C[M, ldC]  =  concat(A1 | gather(A2)) @ B  (+Cin)(+bias)(ReLU)
//
// Block = 256 threads (8 waves) -> 128 rows x (CT*16) cols per blockIdx.y.
// B panel (K x CT*16) staged once per block into LDS as f16, column-major with
// +8-half padding, so each lane fragment = two aligned ds_load_b128.
// A fragments = four global b128 loads per k-chunk (two contiguous 8-float
// runs per lane; runs never straddle the A1/A2 concat boundary since K1 and
// run starts are multiples of 8).  All epilogue options are compile-time.
// ---------------------------------------------------------------------------
#define KMAX  256
#define KPAD  8
#define KPMAX (KMAX + KPAD)

template <int CT, bool HAS_A2, bool HAS_CIN, bool HAS_BIAS, bool DO_RELU>
__global__ void wmma_gemm_kernel(const float* __restrict__ A1, int K1,
                                 const float* __restrict__ A2, int K2,
                                 const int* __restrict__ gidx,
                                 const float* __restrict__ B, int ldB,
                                 const float* __restrict__ Cin,
                                 const float* __restrict__ bias,
                                 float* __restrict__ C, int ldC, int M) {
  __shared__ __align__(16) _Float16 Bt[CT * 16 * KPMAX];

  const int lane = threadIdx.x & 31;
  const int wave = threadIdx.x >> 5;
  const int row0 = blockIdx.x * 128 + wave * 16;
  const int colBase = blockIdx.y * (CT * 16);
  const int K  = HAS_A2 ? (K1 + K2) : K1;
  const int KP = K + KPAD;
  const int NC = CT * 16;

  // ---- cooperative stage of B panel: f32 row-major -> f16 column-major LDS
  for (int i = threadIdx.x; i < NC * K; i += 256) {
    int nc = i % NC;        // consecutive threads -> consecutive cols (coalesced)
    int k  = i / NC;
    Bt[nc * KP + k] = (_Float16)B[(size_t)k * ldB + colBase + nc];
  }
  __syncthreads();

  if (row0 < M) {
    const int mrow  = min(row0 + (lane & 15), M - 1);
    const int abase = (lane < 16) ? 0 : 8;     // A: K split across lane halves
    const int kb    = (lane < 16) ? 0 : 8;     // B: K split across lane halves
    const int ncol  = lane & 15;

    const float* rowA1 = A1 + (size_t)mrow * K1;
    const float* rowA2 = nullptr;
    if (HAS_A2) {
      int g = gidx ? gidx[mrow] : mrow;
      rowA2 = A2 + (size_t)g * K2;
    }

    v8f acc[CT];
    v8f zero = {};
#pragma unroll
    for (int t = 0; t < CT; ++t) acc[t] = zero;

    for (int k0 = 0; k0 < K; k0 += 32) {
      // two 8-float runs: K = [k0+abase, +8) and [k0+abase+16, +8)
      int kkl = k0 + abase;
      int kkh = k0 + abase + 16;
      const float* pl;
      const float* ph;
      if (HAS_A2) {
        pl = (kkl < K1) ? (rowA1 + kkl) : (rowA2 + (kkl - K1));
        ph = (kkh < K1) ? (rowA1 + kkh) : (rowA2 + (kkh - K1));
      } else {
        pl = rowA1 + kkl;
        ph = rowA1 + kkh;
      }
      float4 f0 = *(const float4*)(pl);
      float4 f1 = *(const float4*)(pl + 4);
      float4 f2 = *(const float4*)(ph);
      float4 f3 = *(const float4*)(ph + 4);
      v16h a;
      a[0]  = (_Float16)f0.x; a[1]  = (_Float16)f0.y;
      a[2]  = (_Float16)f0.z; a[3]  = (_Float16)f0.w;
      a[4]  = (_Float16)f1.x; a[5]  = (_Float16)f1.y;
      a[6]  = (_Float16)f1.z; a[7]  = (_Float16)f1.w;
      a[8]  = (_Float16)f2.x; a[9]  = (_Float16)f2.y;
      a[10] = (_Float16)f2.z; a[11] = (_Float16)f2.w;
      a[12] = (_Float16)f3.x; a[13] = (_Float16)f3.y;
      a[14] = (_Float16)f3.z; a[15] = (_Float16)f3.w;

#pragma unroll
      for (int t = 0; t < CT; ++t) {
        const _Float16* bp = Bt + (t * 16 + ncol) * KP + k0 + kb;
        v8h blo = *(const v8h*)(bp);
        v8h bhi = *(const v8h*)(bp + 16);
        v16h b = __builtin_shufflevector(blo, bhi, 0, 1, 2, 3, 4, 5, 6, 7,
                                         8, 9, 10, 11, 12, 13, 14, 15);
        acc[t] = __builtin_amdgcn_wmma_f32_16x16x32_f16(
            false, a, false, b, (short)0, acc[t], false, false);
      }
    }

    const int mbase = row0 + ((lane < 16) ? 0 : 8);
#pragma unroll
    for (int t = 0; t < CT; ++t) {
      const int n = colBase + t * 16 + ncol;
      float bv = 0.0f;
      if (HAS_BIAS) bv = bias[n];
#pragma unroll
      for (int r = 0; r < 8; ++r) {
        int m = mbase + r;
        if (m < M) {
          float v = acc[t][r];
          if (HAS_CIN) v += Cin[(size_t)m * ldC + n];
          if (HAS_BIAS) v += bv;
          if (DO_RELU) v = v > 0.0f ? v : 0.0f;
          C[(size_t)m * ldC + n] = v;
        }
      }
    }
  }
}

// ---------------------------------------------------------------------------
// GCN edge aggregation: h1[dst] += hW[src] * dinv[src]*dinv[dst]
// ---------------------------------------------------------------------------
__global__ void gcn_edge_agg(const int* __restrict__ src, const int* __restrict__ dst,
                             const float* __restrict__ hW, const float* __restrict__ dinv,
                             float* __restrict__ out, long long E) {
  long long t = (long long)blockIdx.x * blockDim.x + threadIdx.x;
  if (t >= E * 32) return;
  long long e = t >> 5;
  int c0 = (int)(t & 31) * 4;
  int s = src[e], d = dst[e];
  float nrm = dinv[s] * dinv[d];
  __builtin_prefetch(out + (size_t)d * D_H + c0, 1, 0);
  const float4 v = *(const float4*)(hW + (size_t)s * D_H + c0);
  float* o = out + (size_t)d * D_H + c0;
  atomicAdd(o + 0, v.x * nrm); atomicAdd(o + 1, v.y * nrm);
  atomicAdd(o + 2, v.z * nrm); atomicAdd(o + 3, v.w * nrm);
}

// self-loop term + bias + relu
__global__ void gcn_epilogue(float* __restrict__ h1, const float* __restrict__ hW,
                             const float* __restrict__ dinv,
                             const float* __restrict__ b_gcn, long long N) {
  long long t = (long long)blockIdx.x * blockDim.x + threadIdx.x;
  if (t >= N * D_H) return;
  long long n = t >> 7;
  int c = (int)(t & 127);
  float di = dinv[n];
  float v = h1[t] + hW[t] * di * di + b_gcn[c];
  h1[t] = v > 0.0f ? v : 0.0f;
}

// ---------------------------------------------------------------------------
// SAGE mean aggregation
// ---------------------------------------------------------------------------
__global__ void sage_edge_agg(const int* __restrict__ src, const int* __restrict__ dst,
                              const float* __restrict__ h, float* __restrict__ agg,
                              long long E) {
  long long t = (long long)blockIdx.x * blockDim.x + threadIdx.x;
  if (t >= E * 32) return;
  long long e = t >> 5;
  int c0 = (int)(t & 31) * 4;
  int s = src[e], d = dst[e];
  const float4 v = *(const float4*)(h + (size_t)s * D_H + c0);
  float* o = agg + (size_t)d * D_H + c0;
  atomicAdd(o + 0, v.x); atomicAdd(o + 1, v.y);
  atomicAdd(o + 2, v.z); atomicAdd(o + 3, v.w);
}

__global__ void sage_mean(float* __restrict__ agg, const float* __restrict__ deg,
                          long long N) {
  long long t = (long long)blockIdx.x * blockDim.x + threadIdx.x;
  if (t >= N * D_H) return;
  long long n = t >> 7;
  agg[t] /= fmaxf(deg[n], 1.0f);
}

// ---------------------------------------------------------------------------
// GAT
// ---------------------------------------------------------------------------
__global__ void gat_att_dots(const float* __restrict__ xW,
                             const float* __restrict__ att_src,
                             const float* __restrict__ att_dst,
                             float* __restrict__ a_src, float* __restrict__ a_dst,
                             long long N) {
  long long t = (long long)blockIdx.x * blockDim.x + threadIdx.x;
  if (t >= N * HEADS) return;
  long long n = t >> 1;
  int h = (int)(t & 1);
  const float* xw = xW + (size_t)n * (HEADS * D_H) + (size_t)h * D_H;
  const float* as = att_src + (size_t)h * D_H;
  const float* ad = att_dst + (size_t)h * D_H;
  float s1 = 0.0f, s2 = 0.0f;
#pragma unroll 4
  for (int c = 0; c < D_H; c += 4) {
    float4 x = *(const float4*)(xw + c);
    float4 a = *(const float4*)(as + c);
    float4 b = *(const float4*)(ad + c);
    s1 += x.x * a.x + x.y * a.y + x.z * a.z + x.w * a.w;
    s2 += x.x * b.x + x.y * b.y + x.z * b.z + x.w * b.w;
  }
  a_src[t] = s1;
  a_dst[t] = s2;
}

__device__ __forceinline__ unsigned f2ord(float f) {
  unsigned u = __float_as_uint(f);
  return (u & 0x80000000u) ? ~u : (u | 0x80000000u);
}
__device__ __forceinline__ float ord2f(unsigned k) {
  return (k & 0x80000000u) ? __uint_as_float(k & 0x7fffffffu) : __uint_as_float(~k);
}
__device__ __forceinline__ float lrelu02(float v) { return v > 0.0f ? v : 0.2f * v; }

// edge set = E real edges + N self loops (i in [E, E+N) -> src=dst=i-E)
__global__ void gat_max_pass(const int* __restrict__ src, const int* __restrict__ dst,
                             const float* __restrict__ a_src, const float* __restrict__ a_dst,
                             unsigned* __restrict__ amax, long long E, long long N) {
  long long t = (long long)blockIdx.x * blockDim.x + threadIdx.x;
  if (t >= (E + N) * HEADS) return;
  long long i = t >> 1;
  int h = (int)(t & 1);
  int s, d;
  if (i < E) { s = src[i]; d = dst[i]; } else { s = d = (int)(i - E); }
  float v = lrelu02(a_src[(size_t)s * HEADS + h] + a_dst[(size_t)d * HEADS + h]);
  atomicMax(&amax[(size_t)d * HEADS + h], f2ord(v));
}

__global__ void gat_sum_pass(const int* __restrict__ src, const int* __restrict__ dst,
                             const float* __restrict__ a_src, const float* __restrict__ a_dst,
                             const unsigned* __restrict__ amax, float* __restrict__ asum,
                             long long E, long long N) {
  long long t = (long long)blockIdx.x * blockDim.x + threadIdx.x;
  if (t >= (E + N) * HEADS) return;
  long long i = t >> 1;
  int h = (int)(t & 1);
  int s, d;
  if (i < E) { s = src[i]; d = dst[i]; } else { s = d = (int)(i - E); }
  float v = lrelu02(a_src[(size_t)s * HEADS + h] + a_dst[(size_t)d * HEADS + h]);
  float alpha = expf(v - ord2f(amax[(size_t)d * HEADS + h]));
  atomicAdd(&asum[(size_t)d * HEADS + h], alpha);
}

// one thread = one float4 chunk of one (edge, head)
__global__ void gat_scatter(const int* __restrict__ src, const int* __restrict__ dst,
                            const float* __restrict__ a_src, const float* __restrict__ a_dst,
                            const unsigned* __restrict__ amax, const float* __restrict__ asum,
                            const float* __restrict__ xW, float* __restrict__ gout,
                            long long E, long long N) {
  long long t = (long long)blockIdx.x * blockDim.x + threadIdx.x;
  if (t >= (E + N) * 64) return;
  long long i = t >> 6;
  int r = (int)(t & 63);
  int h = r >> 5;
  int c0 = (r & 31) * 4;
  int s, d;
  if (i < E) { s = src[i]; d = dst[i]; } else { s = d = (int)(i - E); }
  float v = lrelu02(a_src[(size_t)s * HEADS + h] + a_dst[(size_t)d * HEADS + h]);
  float alpha = expf(v - ord2f(amax[(size_t)d * HEADS + h]));
  float coef = alpha / (asum[(size_t)d * HEADS + h] + 1e-16f);
  const float4 x = *(const float4*)(xW + (size_t)s * (HEADS * D_H) + (size_t)h * D_H + c0);
  float* o = gout + (size_t)d * (HEADS * D_H) + (size_t)h * D_H + c0;
  atomicAdd(o + 0, x.x * coef); atomicAdd(o + 1, x.y * coef);
  atomicAdd(o + 2, x.z * coef); atomicAdd(o + 3, x.w * coef);
}

// mean over heads + bias + relu
__global__ void gat_epilogue(const float* __restrict__ gout, const float* __restrict__ b_gat,
                             float* __restrict__ h3, long long N) {
  long long t = (long long)blockIdx.x * blockDim.x + threadIdx.x;
  if (t >= N * D_H) return;
  long long n = t >> 7;
  int c = (int)(t & 127);
  float v = 0.5f * (gout[(size_t)n * 256 + c] + gout[(size_t)n * 256 + 128 + c]) + b_gat[c];
  h3[t] = v > 0.0f ? v : 0.0f;
}

// ---------------------------------------------------------------------------
// launch
// ---------------------------------------------------------------------------
static inline unsigned blocks_for(long long n, int bs) {
  return (unsigned)((n + bs - 1) / bs);
}

extern "C" void kernel_launch(void* const* d_in, const int* in_sizes, int n_in,
                              void* d_out, int out_size, void* d_ws, size_t ws_size,
                              hipStream_t stream) {
  const float* x        = (const float*)d_in[0];
  const int*   edge     = (const int*)d_in[1];
  const int*   nidx     = (const int*)d_in[2];
  const float* emb      = (const float*)d_in[3];
  const float* W_gcn    = (const float*)d_in[4];
  const float* b_gcn    = (const float*)d_in[5];
  const float* W_sage_l = (const float*)d_in[6];
  const float* W_sage_r = (const float*)d_in[7];
  const float* b_sage   = (const float*)d_in[8];
  const float* W_gat    = (const float*)d_in[9];   // [128,2,128] == row-major [128,256]
  const float* att_src  = (const float*)d_in[10];
  const float* att_dst  = (const float*)d_in[11];
  const float* b_gat    = (const float*)d_in[12];
  const float* W_out    = (const float*)d_in[13];
  const float* b_out    = (const float*)d_in[14];
  float* out = (float*)d_out;

  const long long N = in_sizes[2];
  const long long E = in_sizes[1] / 2;
  const int* esrc = edge;
  const int* edst = edge + E;

  // workspace layout (floats), each region 16B aligned
  float* ws = (float*)d_ws;
  size_t off = 0;
  auto take = [&](size_t n) { float* p = ws + off; off += (n + 3) & ~(size_t)3; return p; };
  float*    dinv  = take(N);
  float*    deg   = take(N);
  float*    a_src = take(2 * N);
  float*    a_dst = take(2 * N);
  unsigned* amax  = (unsigned*)take(2 * N);
  float*    asum  = take(2 * N);
  float*    hW    = take(128 * (size_t)N);   // later reused as SAGE agg/mean
  float*    h1    = take(128 * (size_t)N);   // GCN out; later reused as GAT out h3
  float*    h2    = take(128 * (size_t)N);   // SAGE out
  float*    xW    = take(256 * (size_t)N);   // GAT per-head features
  float*    gout  = take(256 * (size_t)N);   // GAT aggregation

  const int BS = 256;
  dim3 gemm_blk(256);

  // ---- degrees / norm
  fill_f32<<<blocks_for(N, BS), BS, 0, stream>>>(deg, 0.0f, (size_t)N);
  fill_f32<<<blocks_for(N * 128, BS), BS, 0, stream>>>(h1, 0.0f, (size_t)N * 128);
  deg_kernel<<<blocks_for(E, BS), BS, 0, stream>>>(edst, deg, E);
  dinv_kernel<<<blocks_for(N, BS), BS, 0, stream>>>(deg, dinv, N);

  // ---- GEMM1: hW = [x | emb[nidx]] @ W_gcn        (M=N, K=256, Dout=128)
  {
    dim3 grid(blocks_for(N, 128), 2);
    wmma_gemm_kernel<4, true, false, false, false><<<grid, gemm_blk, 0, stream>>>(
        x, 128, emb, 128, nidx, W_gcn, 128, nullptr, nullptr, hW, 128, (int)N);
  }

  // ---- GCN aggregation + epilogue
  gcn_edge_agg<<<blocks_for(E * 32, BS), BS, 0, stream>>>(esrc, edst, hW, dinv, h1, E);
  gcn_epilogue<<<blocks_for(N * 128, BS), BS, 0, stream>>>(h1, hW, dinv, b_gcn, N);

  // ---- SAGE: agg (reuse hW buffer), mean, two GEMMs into h2
  fill_f32<<<blocks_for(N * 128, BS), BS, 0, stream>>>(hW, 0.0f, (size_t)N * 128);
  sage_edge_agg<<<blocks_for(E * 32, BS), BS, 0, stream>>>(esrc, edst, h1, hW, E);
  sage_mean<<<blocks_for(N * 128, BS), BS, 0, stream>>>(hW, deg, N);
  {
    dim3 grid(blocks_for(N, 128), 2);
    wmma_gemm_kernel<4, false, false, false, false><<<grid, gemm_blk, 0, stream>>>(
        hW, 128, nullptr, 0, nullptr, W_sage_l, 128, nullptr, nullptr, h2, 128, (int)N);
    wmma_gemm_kernel<4, false, true, true, true><<<grid, gemm_blk, 0, stream>>>(
        h1, 128, nullptr, 0, nullptr, W_sage_r, 128, h2, b_sage, h2, 128, (int)N);
  }

  // ---- GAT: xW = h2 @ W_gat  (Dout=256)
  {
    dim3 grid(blocks_for(N, 128), 4);
    wmma_gemm_kernel<4, false, false, false, false><<<grid, gemm_blk, 0, stream>>>(
        h2, 128, nullptr, 0, nullptr, W_gat, 256, nullptr, nullptr, xW, 256, (int)N);
  }
  gat_att_dots<<<blocks_for(N * HEADS, BS), BS, 0, stream>>>(xW, att_src, att_dst,
                                                            a_src, a_dst, N);
  fill_f32<<<blocks_for(N * 2, BS), BS, 0, stream>>>((float*)amax, 0.0f, (size_t)N * 2);
  fill_f32<<<blocks_for(N * 2, BS), BS, 0, stream>>>(asum, 0.0f, (size_t)N * 2);
  gat_max_pass<<<blocks_for((E + N) * HEADS, BS), BS, 0, stream>>>(
      esrc, edst, a_src, a_dst, amax, E, N);
  gat_sum_pass<<<blocks_for((E + N) * HEADS, BS), BS, 0, stream>>>(
      esrc, edst, a_src, a_dst, amax, asum, E, N);
  fill_f32<<<blocks_for(N * 256, BS), BS, 0, stream>>>(gout, 0.0f, (size_t)N * 256);
  gat_scatter<<<blocks_for((E + N) * 64, BS), BS, 0, stream>>>(
      esrc, edst, a_src, a_dst, amax, asum, xW, gout, E, N);
  gat_epilogue<<<blocks_for(N * 128, BS), BS, 0, stream>>>(gout, b_gat, h1, N);

  // ---- out = h3 @ W_out + b_out   (Dout=64)
  {
    dim3 grid(blocks_for(N, 128), 1);
    wmma_gemm_kernel<4, false, false, true, false><<<grid, gemm_blk, 0, stream>>>(
        h1, 128, nullptr, 0, nullptr, W_out, 64, nullptr, b_out, out, 64, (int)N);
  }
}
